// PatchCoreONNXWrapper_24799141167279
// MI455X (gfx1250) — matrix operand order repaired and checked by reading, not compiled, not run on hardware
//
#include <hip/hip_runtime.h>

// ---------------------------------------------------------------------------
// Problem sizes (fixed by the reference)
// ---------------------------------------------------------------------------
#define N_PATCH 12544      // 16 * 28 * 28
#define K_DIM   1536       // 512 + 1024
#define M_MEM   16384
#define B_IMG   16
#define H_FEAT  28
#define W_FEAT  28
#define H_OUT   224
#define W_OUT   224
#define MAP_ELEMS (B_IMG * H_OUT * W_OUT)   // 802816

// ---------------------------------------------------------------------------
// Types for CDNA5 WMMA
// ---------------------------------------------------------------------------
typedef __bf16 bf16_t;
typedef bf16_t v16bf __attribute__((ext_vector_type(16)));
typedef bf16_t v8bf  __attribute__((ext_vector_type(8)));
typedef float  v8f   __attribute__((ext_vector_type(8)));

static __device__ __forceinline__ unsigned short f2bf(float f) {
    unsigned u = __float_as_uint(f);
    unsigned r = u + 0x7FFFu + ((u >> 16) & 1u);   // round-to-nearest-even
    return (unsigned short)(r >> 16);
}

// Async global -> LDS copy of 16 bytes (CDNA5 GLOBAL_LOAD_ASYNC_TO_LDS_B128,
// tracked by ASYNCcnt). vdst = LDS byte offset (low 32 bits of shared addr),
// vaddr = 64-bit global address.
static __device__ __forceinline__ void async_copy_b128(void* lds_ptr,
                                                       const void* gptr) {
    unsigned lds_off = (unsigned)(unsigned long long)lds_ptr;
    unsigned long long ga = (unsigned long long)gptr;
    asm volatile("global_load_async_to_lds_b128 %0, %1, off"
                 :: "v"(lds_off), "v"(ga)
                 : "memory");
}

static __device__ __forceinline__ void wait_async0() {
#if __has_builtin(__builtin_amdgcn_s_wait_asynccnt)
    __builtin_amdgcn_s_wait_asynccnt(0);
#else
    asm volatile("s_wait_asynccnt 0x0" ::: "memory");
#endif
}

// ---------------------------------------------------------------------------
// Kernel 1: build bf16 feature rows (concat feat2 with bilinear-2x feat3)
// and per-row squared norm. One block per patch.
// ---------------------------------------------------------------------------
__global__ __launch_bounds__(256)
void build_feats(const float* __restrict__ feat2,
                 const float* __restrict__ feat3,
                 unsigned short* __restrict__ featsb,
                 float* __restrict__ xsq) {
    const int n = blockIdx.x;                 // 0..12543
    const int b = n / (H_FEAT * W_FEAT);
    const int rem = n % (H_FEAT * W_FEAT);
    const int y = rem / W_FEAT;
    const int x = rem % W_FEAT;

    // half-pixel bilinear source coords for 14 -> 28 upsample
    const float fy = 0.5f * (float)y - 0.25f;
    const float fx = 0.5f * (float)x - 0.25f;
    const int y0 = (int)floorf(fy), x0 = (int)floorf(fx);
    const float wy = fy - (float)y0, wx = fx - (float)x0;
    const int ya = min(max(y0, 0), 13),     yb = min(max(y0 + 1, 0), 13);
    const int xa = min(max(x0, 0), 13),     xb = min(max(x0 + 1, 0), 13);
    const float w00 = (1.f - wy) * (1.f - wx), w01 = (1.f - wy) * wx;
    const float w10 = wy * (1.f - wx),         w11 = wy * wx;

    float ss = 0.f;
    for (int c = threadIdx.x; c < K_DIM; c += 256) {
        float v;
        if (c < 512) {
            v = feat2[((size_t)(b * 512 + c) * H_FEAT + y) * W_FEAT + x];
        } else {
            const int c3 = c - 512;
            const float* p = feat3 + (size_t)(b * 1024 + c3) * 196;
            v = w00 * p[ya * 14 + xa] + w01 * p[ya * 14 + xb] +
                w10 * p[yb * 14 + xa] + w11 * p[yb * 14 + xb];
        }
        ss += v * v;
        featsb[(size_t)n * K_DIM + c] = f2bf(v);
    }
    __shared__ float red[256];
    red[threadIdx.x] = ss;
    __syncthreads();
    for (int s = 128; s > 0; s >>= 1) {
        if (threadIdx.x < s) red[threadIdx.x] += red[threadIdx.x + s];
        __syncthreads();
    }
    if (threadIdx.x == 0) xsq[n] = red[0];
}

// ---------------------------------------------------------------------------
// Kernel 2: memory bank fp32 -> bf16 + squared norms. One block per row.
// ---------------------------------------------------------------------------
__global__ __launch_bounds__(256)
void conv_bank(const float* __restrict__ mbank,
               unsigned short* __restrict__ mbankb,
               float* __restrict__ msq) {
    const int m = blockIdx.x;
    float ss = 0.f;
    for (int k = threadIdx.x; k < K_DIM; k += 256) {
        float v = mbank[(size_t)m * K_DIM + k];
        ss += v * v;
        mbankb[(size_t)m * K_DIM + k] = f2bf(v);
    }
    __shared__ float red[256];
    red[threadIdx.x] = ss;
    __syncthreads();
    for (int s = 128; s > 0; s >>= 1) {
        if (threadIdx.x < s) red[threadIdx.x] += red[threadIdx.x + s];
        __syncthreads();
    }
    if (threadIdx.x == 0) msq[m] = red[0];
}

// ---------------------------------------------------------------------------
// Kernel 3: init per-row min (fp32 bits, +inf)
// ---------------------------------------------------------------------------
__global__ void init_min(unsigned* __restrict__ gmin) {
    int i = blockIdx.x * 256 + threadIdx.x;
    if (i < N_PATCH) gmin[i] = 0x7F800000u;
}

// ---------------------------------------------------------------------------
// Kernel 4: bf16 WMMA GEMM (X * M^T) fused with distance + row-min.
//   Block: 256 threads = 8 waves, output tile 128(rows) x 128(cols), BK=64.
//   Wave (2x4 grid) owns 64x32 -> 4x2 tiles of v_wmma_f32_16x16x32_bf16.
//   Tiles stream global->LDS through the CDNA5 async-to-LDS path with a
//   double-buffered software pipeline (ASYNCcnt + workgroup barrier).
// ---------------------------------------------------------------------------
__global__ __launch_bounds__(256)
void gemm_min(const unsigned short* __restrict__ A,   // feats bf16 [N, K]
              const unsigned short* __restrict__ B,   // mbank bf16 [M, K]
              const float* __restrict__ xsq,
              const float* __restrict__ msq,
              unsigned* __restrict__ gmin) {
    __shared__ __align__(32) unsigned short sA[2][128 * 64];
    __shared__ __align__(32) unsigned short sB[2][128 * 64];
    __shared__ unsigned sMin[128];

    const int tid  = threadIdx.x;
    const int lane = tid & 31;
    const int wid  = tid >> 5;
    const int wrow = wid >> 2;        // 0..1  -> 64-row slab
    const int wcol = wid & 3;         // 0..3  -> 32-col slab
    const int h    = lane >> 4;       // K-half selector (CDNA5 bf16 layout)
    const int l15  = lane & 15;

    const int m0 = blockIdx.x * 128;  // memory-bank cols
    const int n0 = blockIdx.y * 128;  // patch rows

    if (tid < 128) sMin[tid] = 0x7F800000u;

    // Issue async loads for a 128x64 bf16 slab of A and B into buffer `bsel`.
    auto issue_tile = [&](int k0, int bsel) {
        #pragma unroll
        for (int i = 0; i < 4; i++) {
            const int chunk = tid + 256 * i;      // 0..1023 16B-chunks
            const int row  = chunk >> 3;          // 8 chunks per 64-elem row
            const int c16  = chunk & 7;
            async_copy_b128(&sA[bsel][chunk * 8],
                            A + (size_t)(n0 + row) * K_DIM + k0 + c16 * 8);
            async_copy_b128(&sB[bsel][chunk * 8],
                            B + (size_t)(m0 + row) * K_DIM + k0 + c16 * 8);
        }
    };

    v8f acc[4][2];
    #pragma unroll
    for (int r = 0; r < 4; r++)
        #pragma unroll
        for (int c = 0; c < 2; c++)
            #pragma unroll
            for (int i = 0; i < 8; i++) acc[r][c][i] = 0.0f;

    issue_tile(0, 0);                 // prologue: fill buffer 0

    int buf = 0;
    for (int k0 = 0; k0 < K_DIM; k0 += 64) {
        // Wait for this wave's async copies into `buf`, then sync the
        // workgroup so every wave's LDS writes are visible.
        wait_async0();
        __syncthreads();

        if (k0 + 64 < K_DIM) issue_tile(k0 + 64, buf ^ 1);   // overlap next slab

        const unsigned short* tA = sA[buf];
        const unsigned short* tB = sB[buf];

        #pragma unroll
        for (int kk = 0; kk < 2; kk++) {
            // Hoist all fragments so the 8 WMMAs issue back-to-back.
            v16bf bfrag[2];
            #pragma unroll
            for (int c = 0; c < 2; c++) {
                // B lane layout: col = l15, K-run = kk*32 + h*16 .. +15 (32B)
                bfrag[c] = *(const v16bf*)
                    &tB[(size_t)(wcol * 32 + c * 16 + l15) * 64 + kk * 32 + h * 16];
            }
            v16bf afrag[4];
            #pragma unroll
            for (int r = 0; r < 4; r++) {
                // A lane layout: row = l15, two 8-elem runs at kk*32+h*16 / +8
                const unsigned short* pa =
                    &tA[(size_t)(wrow * 64 + r * 16 + l15) * 64 + kk * 32 + h * 16];
                v8bf lo = *(const v8bf*)pa;
                v8bf hi = *(const v8bf*)(pa + 8);
                #pragma unroll
                for (int i = 0; i < 8; i++) { afrag[r][i] = lo[i]; afrag[r][i + 8] = hi[i]; }
            }
            #pragma unroll
            for (int r = 0; r < 4; r++)
                #pragma unroll
                for (int c = 0; c < 2; c++)
                    acc[r][c] = __builtin_amdgcn_wmma_f32_16x16x32_bf16(
                        false, afrag[r], false, bfrag[c], (short)0, acc[r][c],
                        false, false);
        }
        buf ^= 1;
    }

    // ---- epilogue: d2 = x^2 + m^2 - 2*dot ; min over this block's 128 cols ----
    const float ms0 = msq[m0 + wcol * 32 + l15];
    const float ms1 = msq[m0 + wcol * 32 + 16 + l15];
    #pragma unroll
    for (int r = 0; r < 4; r++) {
        #pragma unroll
        for (int v = 0; v < 8; v++) {
            const int row = wrow * 64 + r * 16 + h * 8 + v;   // C layout: M = 8*h + v
            const float xs = xsq[n0 + row];
            const float d0 = xs + ms0 - 2.0f * acc[r][0][v];
            const float d1 = xs + ms1 - 2.0f * acc[r][1][v];
            float d = fminf(d0, d1);
            d = fmaxf(d, 0.0f);
            // min across the 16 lanes of this half-wave (same row, 16 cols each)
            #pragma unroll
            for (int mshf = 1; mshf < 16; mshf <<= 1)
                d = fminf(d, __shfl_xor(d, mshf, 32));
            if (l15 == 0) atomicMin(&sMin[row], __float_as_uint(d));
        }
    }
    __syncthreads();
    if (tid < 128) atomicMin(&gmin[n0 + tid], sMin[tid]);
}

// ---------------------------------------------------------------------------
// Kernel 5: min_d = sqrt(min_d2)
// ---------------------------------------------------------------------------
__global__ void sqrt_min(const unsigned* __restrict__ gmin,
                         float* __restrict__ mind) {
    int i = blockIdx.x * 256 + threadIdx.x;
    if (i < N_PATCH) mind[i] = sqrtf(__uint_as_float(gmin[i]));
}

// ---------------------------------------------------------------------------
// Kernel 6: bilinear 8x upsample 28x28 -> 224x224 (half-pixel)
// ---------------------------------------------------------------------------
__global__ __launch_bounds__(256)
void upsample8x(const float* __restrict__ mind, float* __restrict__ out) {
    const int idx = blockIdx.x * 256 + threadIdx.x;     // 0..802815
    if (idx >= MAP_ELEMS) return;
    const int b  = idx / (H_OUT * W_OUT);
    const int r  = idx % (H_OUT * W_OUT);
    const int oy = r / W_OUT;
    const int ox = r % W_OUT;

    const float fy = (float)oy * 0.125f - 0.4375f;
    const float fx = (float)ox * 0.125f - 0.4375f;
    const int y0 = (int)floorf(fy), x0 = (int)floorf(fx);
    const float wy = fy - (float)y0, wx = fx - (float)x0;
    const int ya = min(max(y0, 0), 27),     yb = min(max(y0 + 1, 0), 27);
    const int xa = min(max(x0, 0), 27),     xb = min(max(x0 + 1, 0), 27);

    const float* p = mind + (size_t)b * (H_FEAT * W_FEAT);
    const float v =
        (1.f - wy) * ((1.f - wx) * p[ya * 28 + xa] + wx * p[ya * 28 + xb]) +
        wy         * ((1.f - wx) * p[yb * 28 + xa] + wx * p[yb * 28 + xb]);
    out[idx] = v;
}

// ---------------------------------------------------------------------------
// Kernel 7: per-image max over the 224x224 map -> pred_score
// ---------------------------------------------------------------------------
__global__ __launch_bounds__(256)
void image_max(const float* __restrict__ map, float* __restrict__ score) {
    const int b = blockIdx.x;
    float mx = -3.402823466e+38f;
    for (int i = threadIdx.x; i < H_OUT * W_OUT; i += 256)
        mx = fmaxf(mx, map[(size_t)b * (H_OUT * W_OUT) + i]);
    __shared__ float red[256];
    red[threadIdx.x] = mx;
    __syncthreads();
    for (int s = 128; s > 0; s >>= 1) {
        if (threadIdx.x < s)
            red[threadIdx.x] = fmaxf(red[threadIdx.x], red[threadIdx.x + s]);
        __syncthreads();
    }
    if (threadIdx.x == 0) score[b] = red[0];
}

// ---------------------------------------------------------------------------
// Launcher
// ---------------------------------------------------------------------------
extern "C" void kernel_launch(void* const* d_in, const int* in_sizes, int n_in,
                              void* d_out, int out_size, void* d_ws, size_t ws_size,
                              hipStream_t stream) {
    const float* feat2 = (const float*)d_in[0];   // [16,512,28,28]
    const float* feat3 = (const float*)d_in[1];   // [16,1024,14,14]
    const float* mbank = (const float*)d_in[2];   // [16384,1536]
    float* out = (float*)d_out;                   // 802816 map + 16 scores

    // workspace layout (all offsets 256B-aligned)
    char* ws = (char*)d_ws;
    const size_t OFF_FEATS = 0;                                   // 38,535,168 B
    const size_t OFF_MBANK = OFF_FEATS + (size_t)N_PATCH * K_DIM * 2;
    const size_t OFF_XSQ   = OFF_MBANK + (size_t)M_MEM  * K_DIM * 2;
    const size_t OFF_MSQ   = OFF_XSQ   + (size_t)N_PATCH * 4;
    const size_t OFF_MIND2 = OFF_MSQ   + (size_t)M_MEM  * 4;
    const size_t OFF_MIND  = OFF_MIND2 + (size_t)N_PATCH * 4;

    unsigned short* featsb = (unsigned short*)(ws + OFF_FEATS);
    unsigned short* mbankb = (unsigned short*)(ws + OFF_MBANK);
    float*    xsq   = (float*)(ws + OFF_XSQ);
    float*    msq   = (float*)(ws + OFF_MSQ);
    unsigned* gmin  = (unsigned*)(ws + OFF_MIND2);
    float*    mind  = (float*)(ws + OFF_MIND);

    build_feats<<<N_PATCH, 256, 0, stream>>>(feat2, feat3, featsb, xsq);
    conv_bank<<<M_MEM, 256, 0, stream>>>(mbank, mbankb, msq);
    init_min<<<(N_PATCH + 255) / 256, 256, 0, stream>>>(gmin);

    dim3 grid(M_MEM / 128, N_PATCH / 128);   // 128 x 98
    gemm_min<<<grid, 256, 0, stream>>>(featsb, mbankb, xsq, msq, gmin);

    sqrt_min<<<(N_PATCH + 255) / 256, 256, 0, stream>>>(gmin, mind);
    upsample8x<<<MAP_ELEMS / 256, 256, 0, stream>>>(mind, out);
    image_max<<<B_IMG, 256, 0, stream>>>(out, out + MAP_ELEMS);
}
